// OutputModule_68650757259664
// MI455X (gfx1250) — matrix operand order repaired and checked by reading, not compiled, not run on hardware
//
#include <hip/hip_runtime.h>
#include <cstdint>

typedef __attribute__((ext_vector_type(16))) __bf16 v16bf;
typedef __attribute__((ext_vector_type(8)))  float  v8f;
typedef __attribute__((ext_vector_type(4)))  unsigned int u32x4;
typedef int i32x4_gcc __attribute__((vector_size(16)));   // matches builtin param type

typedef unsigned short u16;
typedef unsigned int   u32;

#define EMBED 512
#define FF    1024
#define KIN   1536
#define NCAT  2048
#define CHUNK 4096
#define NGAUSS 50
#define KRBF  64
#define MAXR  12.0f

#if __has_builtin(__builtin_amdgcn_global_load_async_to_lds_b128)
#define USE_ASYNC_LDS 1
#endif

// ---------- helpers ----------
__device__ __forceinline__ u16 f2bf(float f) {
    u32 u = __builtin_bit_cast(u32, f);
    u32 r = u + 0x7FFFu + ((u >> 16) & 1u);   // round-to-nearest-even
    return (u16)(r >> 16);
}
__device__ __forceinline__ float bf2f(u16 h) {
    u32 u = ((u32)h) << 16;
    return __builtin_bit_cast(float, u);
}
__device__ __forceinline__ float geluf(float x) {
    const float c0 = 0.7978845608028654f;   // sqrt(2/pi)
    const float c1 = 0.044715f;
    return 0.5f * x * (1.0f + tanhf(c0 * (x + c1 * x * x * x)));
}

union FragCast { struct { u32x4 lo, hi; } s; v16bf v; };
__device__ __forceinline__ v16bf load_frag(const u16* p0, const u16* p1) {
    FragCast u;
    u.s.lo = *(const u32x4*)p0;
    u.s.hi = *(const u32x4*)p1;
    return u.v;
}

#ifdef USE_ASYNC_LDS
// 16B global -> LDS, per-lane, tracked by ASYNCcnt (global_load_async_to_lds_b128)
__device__ __forceinline__ void async_cp16(const u16* g, u16* l) {
    __builtin_amdgcn_global_load_async_to_lds_b128(
        (__attribute__((address_space(1))) i32x4_gcc*)(unsigned long long)(uintptr_t)g,
        (__attribute__((address_space(3))) i32x4_gcc*)(unsigned int)(uintptr_t)l,
        0, 0);
}
__device__ __forceinline__ void wait_async0() {
#if __has_builtin(__builtin_amdgcn_s_wait_asynccnt)
    __builtin_amdgcn_s_wait_asynccnt(0);
#else
    asm volatile("s_wait_asynccnt 0x0" ::: "memory");
#endif
}
#endif

// ---------- prep kernels ----------
// W [K][N] f32  ->  WT [N][K] bf16
__global__ void convT_kernel(const float* __restrict__ W, u16* __restrict__ WT,
                             int K, int N) {
    int idx = blockIdx.x * 256 + threadIdx.x;
    if (idx >= K * N) return;
    int k = idx / N, n = idx % N;
    WT[(size_t)n * K + k] = f2bf(W[idx]);
}

// WrbfT [1024][64] bf16, row n: branch=n>>9, col n'=n&511, K index g;
// value = Wrbf[g][n'] * Wfin[n'] (Wfin folded in), zero-padded g>=50.
__global__ void prep_rbfw_kernel(const float* __restrict__ eW, const float* __restrict__ eF,
                                 const float* __restrict__ fW, const float* __restrict__ fF,
                                 u16* __restrict__ WT) {
    int idx = blockIdx.x * 256 + threadIdx.x;
    if (idx >= 1024 * KRBF) return;
    int n = idx / KRBF, g = idx % KRBF;
    int b = n >> 9, np = n & 511;
    float v = 0.0f;
    if (g < NGAUSS) {
        const float* W = b ? fW : eW;
        const float* F = b ? fF : eF;
        v = W[g * EMBED + np] * F[np];
    }
    WT[idx] = f2bf(v);
}

__global__ void bias_cat_kernel(const float* __restrict__ eb, const float* __restrict__ fb,
                                float* __restrict__ out) {
    int idx = blockIdx.x * 256 + threadIdx.x;
    if (idx >= NCAT) return;
    out[idx] = idx < FF ? eb[idx] : fb[idx - FF];
}

__global__ void zero_kernel(float* __restrict__ p, int n) {
    int idx = blockIdx.x * 256 + threadIdx.x;
    if (idx < n) p[idx] = 0.0f;
}

// ---------- per-chunk kernels ----------
// Build A[CHUNK][1536] bf16 = concat(x[src], x[dst], x[nNodes+e])
// 192 threads/block, thread t handles 8 consecutive cols: 2x b128 load + 1x b128 store
__global__ void gather_kernel(const float* __restrict__ x, const int* __restrict__ ei,
                              u16* __restrict__ A, int ebase, int E, int nNodes) {
    int r = blockIdx.x;
    int e = ebase + r;
    int c0 = threadIdx.x * 8;                  // 0..1528
    u32x4* dst = (u32x4*)(A + (size_t)r * KIN + c0);
    if (e >= E) {
        u32x4 z = {0u, 0u, 0u, 0u};
        *dst = z;
        return;
    }
    const float* srcp;
    if (c0 < EMBED)          srcp = x + (size_t)ei[e]     * EMBED + c0;
    else if (c0 < 2 * EMBED) srcp = x + (size_t)ei[E + e] * EMBED + (c0 - EMBED);
    else                     srcp = x + (size_t)(nNodes + e) * EMBED + (c0 - 2 * EMBED);
    float4 v0 = ((const float4*)srcp)[0];
    float4 v1 = ((const float4*)srcp)[1];
    u32x4 o;
    o.x = (u32)f2bf(v0.x) | ((u32)f2bf(v0.y) << 16);
    o.y = (u32)f2bf(v0.z) | ((u32)f2bf(v0.w) << 16);
    o.z = (u32)f2bf(v1.x) | ((u32)f2bf(v1.y) << 16);
    o.w = (u32)f2bf(v1.z) | ((u32)f2bf(v1.w) << 16);
    *dst = o;
}

// RBF[CHUNK][64] bf16 (gaussian smearing, zero padded)
__global__ void rbf_kernel(const float* __restrict__ dist, u16* __restrict__ RBF,
                           int ebase, int E) {
    int idx = blockIdx.x * 256 + threadIdx.x;
    int r = idx / KRBF, g = idx % KRBF;
    if (r >= CHUNK) return;
    int e = ebase + r;
    float v = 0.0f;
    if (e < E && g < NGAUSS) {
        const float spacing = MAXR / (float)(NGAUSS - 1);
        const float coeff = -0.5f / (spacing * spacing);
        float dx = dist[e] - (float)g * spacing;
        v = __expf(coeff * dx * dx);
    }
    RBF[idx] = f2bf(v);
}

// ---------- WMMA GEMM ----------
// C[M][N](f32 acc) = A[M][K](bf16,row-major) * BT[N][K](bf16)^T
// block 128(M) x 128(N); 8 waves as 4(M) x 2(N); each wave 32x64 via
// 8x v_wmma_f32_16x16x32_bf16 per 32-wide K step.
// Double-buffered LDS; tiles staged with GLOBAL_LOAD_ASYNC_TO_LDS_B128 when
// available (ASYNCcnt), else register-staged; one barrier per K step.
// mode 0: outH = bf16(gelu(acc+bias))
// mode 1: outH = bf16(res + gelu(acc+bias))
// mode 2: outF = acc+bias (f32)
#define BM 128
#define BN 128
#define BK 32
__global__ void __launch_bounds__(256)
gemm_kernel(const u16* __restrict__ A, int lda,
            const u16* __restrict__ BT, int ldb,
            int K, const float* __restrict__ bias, int mode,
            u16* __restrict__ outH, float* __restrict__ outF,
            const u16* __restrict__ res, int ldres, int ldo) {
    __shared__ u16 At[2][BM * BK];   // 2 x 8 KB
    __shared__ u16 Bt[2][BN * BK];   // 2 x 8 KB
    const int tid  = threadIdx.x;
    const int lane = tid & 31;
    const int w    = tid >> 5;
    const int wm   = w & 3;       // 4 wave rows (32 M each)
    const int wn   = w >> 2;      // 2 wave cols (64 N each)
    const int blockM = blockIdx.y * BM;
    const int blockN = blockIdx.x * BN;
    const int ml = lane & 15;     // M/N index within 16
    const int kh = lane >> 4;     // K half select

    const int srow = tid >> 2;    // staging row 0..63 (+64 for 2nd half)
    const int sseg = tid & 3;     // 16B segment within 32-element row

    const u16* Ag = A  + (size_t)blockM * lda;
    const u16* Bg = BT + (size_t)blockN * ldb;

    v8f acc[2][4] = {};

#ifdef USE_ASYNC_LDS
    // preload K-tile 0 into buffer 0 (async DMA to LDS)
    async_cp16(Ag + (size_t)srow * lda + sseg * 8,        At[0] + srow * BK + sseg * 8);
    async_cp16(Ag + (size_t)(srow + 64) * lda + sseg * 8, At[0] + (srow + 64) * BK + sseg * 8);
    async_cp16(Bg + (size_t)srow * ldb + sseg * 8,        Bt[0] + srow * BK + sseg * 8);
    async_cp16(Bg + (size_t)(srow + 64) * ldb + sseg * 8, Bt[0] + (srow + 64) * BK + sseg * 8);
    wait_async0();
    __syncthreads();

    int cur = 0;
    for (int kk = 0; kk < K; kk += BK) {
        const int kn = kk + BK;
        if (kn < K) {   // DMA next tile into the other buffer, overlapping WMMA
            u16* An = At[cur ^ 1];
            u16* Bn = Bt[cur ^ 1];
            async_cp16(Ag + (size_t)srow * lda + kn + sseg * 8,        An + srow * BK + sseg * 8);
            async_cp16(Ag + (size_t)(srow + 64) * lda + kn + sseg * 8, An + (srow + 64) * BK + sseg * 8);
            async_cp16(Bg + (size_t)srow * ldb + kn + sseg * 8,        Bn + srow * BK + sseg * 8);
            async_cp16(Bg + (size_t)(srow + 64) * ldb + kn + sseg * 8, Bn + (srow + 64) * BK + sseg * 8);
            if (kn + BK < K) {   // L2 prefetch of the tile after that
                __builtin_prefetch(Ag + (size_t)srow * lda + kn + BK + sseg * 8, 0, 1);
                __builtin_prefetch(Bg + (size_t)srow * ldb + kn + BK + sseg * 8, 0, 1);
            }
        }

        const u16* Ab = At[cur];
        const u16* Bb = Bt[cur];
        v16bf a[2], b[4];
#pragma unroll
        for (int i = 0; i < 2; ++i) {
            const u16* base = Ab + (wm * 32 + i * 16 + ml) * BK;
            a[i] = load_frag(base + 8 * kh, base + 16 + 8 * kh);
        }
#pragma unroll
        for (int j = 0; j < 4; ++j) {
            const u16* base = Bb + (wn * 64 + j * 16 + ml) * BK;
            b[j] = load_frag(base + 8 * kh, base + 16 + 8 * kh);
        }
#pragma unroll
        for (int i = 0; i < 2; ++i)
#pragma unroll
            for (int j = 0; j < 4; ++j)
                acc[i][j] = __builtin_amdgcn_wmma_f32_16x16x32_bf16(
                    false, a[i], false, b[j], (short)0, acc[i][j], false, false);

        if (kn < K) wait_async0();
        __syncthreads();
        cur ^= 1;
    }
#else
    // fallback: register-staged double buffering
    *((u32x4*)(At[0] + srow * BK + sseg * 8)) =
        *((const u32x4*)(Ag + (size_t)srow * lda + sseg * 8));
    *((u32x4*)(At[0] + (srow + 64) * BK + sseg * 8)) =
        *((const u32x4*)(Ag + (size_t)(srow + 64) * lda + sseg * 8));
    *((u32x4*)(Bt[0] + srow * BK + sseg * 8)) =
        *((const u32x4*)(Bg + (size_t)srow * ldb + sseg * 8));
    *((u32x4*)(Bt[0] + (srow + 64) * BK + sseg * 8)) =
        *((const u32x4*)(Bg + (size_t)(srow + 64) * ldb + sseg * 8));
    __syncthreads();

    int cur = 0;
    for (int kk = 0; kk < K; kk += BK) {
        const int kn = kk + BK;
        u32x4 pa0, pa1, pb0, pb1;
        if (kn < K) {
            pa0 = *((const u32x4*)(Ag + (size_t)srow * lda + kn + sseg * 8));
            pa1 = *((const u32x4*)(Ag + (size_t)(srow + 64) * lda + kn + sseg * 8));
            pb0 = *((const u32x4*)(Bg + (size_t)srow * ldb + kn + sseg * 8));
            pb1 = *((const u32x4*)(Bg + (size_t)(srow + 64) * ldb + kn + sseg * 8));
            if (kn + BK < K) {
                __builtin_prefetch(Ag + (size_t)srow * lda + kn + BK + sseg * 8, 0, 1);
                __builtin_prefetch(Bg + (size_t)srow * ldb + kn + BK + sseg * 8, 0, 1);
            }
        }
        const u16* Ab = At[cur];
        const u16* Bb = Bt[cur];
        v16bf a[2], b[4];
#pragma unroll
        for (int i = 0; i < 2; ++i) {
            const u16* base = Ab + (wm * 32 + i * 16 + ml) * BK;
            a[i] = load_frag(base + 8 * kh, base + 16 + 8 * kh);
        }
#pragma unroll
        for (int j = 0; j < 4; ++j) {
            const u16* base = Bb + (wn * 64 + j * 16 + ml) * BK;
            b[j] = load_frag(base + 8 * kh, base + 16 + 8 * kh);
        }
#pragma unroll
        for (int i = 0; i < 2; ++i)
#pragma unroll
            for (int j = 0; j < 4; ++j)
                acc[i][j] = __builtin_amdgcn_wmma_f32_16x16x32_bf16(
                    false, a[i], false, b[j], (short)0, acc[i][j], false, false);

        if (kn < K) {
            u16* An = At[cur ^ 1];
            u16* Bn = Bt[cur ^ 1];
            *((u32x4*)(An + srow * BK + sseg * 8)) = pa0;
            *((u32x4*)(An + (srow + 64) * BK + sseg * 8)) = pa1;
            *((u32x4*)(Bn + srow * BK + sseg * 8)) = pb0;
            *((u32x4*)(Bn + (srow + 64) * BK + sseg * 8)) = pb1;
        }
        __syncthreads();
        cur ^= 1;
    }
#endif

    // epilogue: lane holds col n = lane&15, rows r + 8*(lane>>4)
#pragma unroll
    for (int i = 0; i < 2; ++i) {
#pragma unroll
        for (int j = 0; j < 4; ++j) {
            int gn = blockN + wn * 64 + j * 16 + ml;
            float bv = bias ? bias[gn] : 0.0f;
            int gm0 = blockM + wm * 32 + i * 16 + 8 * kh;
#pragma unroll
            for (int r = 0; r < 8; ++r) {
                int gm = gm0 + r;
                float v = acc[i][j][r] + bv;
                size_t o = (size_t)gm * ldo + gn;
                if (mode == 0)       outH[o] = f2bf(geluf(v));
                else if (mode == 1)  outH[o] = f2bf(geluf(v) + bf2f(res[(size_t)gm * ldres + gn]));
                else                 outF[o] = v;
            }
        }
    }
}

// ---------- finalize: one wave per edge ----------
__global__ void finalize_kernel(const float* __restrict__ OUTe, const float* __restrict__ OUTf,
                                const float* __restrict__ P,
                                const float* __restrict__ vec_hat,
                                const int* __restrict__ batch, const int* __restrict__ ei,
                                float* __restrict__ energy, float* __restrict__ forces,
                                int ebase, int E) {
    int lane = threadIdx.x & 31;
    int wv   = threadIdx.x >> 5;
    int r    = blockIdx.x * 8 + wv;
    int e    = ebase + r;
    if (e >= E) return;
    const float4* oe4 = (const float4*)(OUTe + (size_t)r * EMBED);
    const float4* of4 = (const float4*)(OUTf + (size_t)r * EMBED);
    const float4* pe4 = (const float4*)(P + (size_t)r * 1024);
    const float4* pf4 = (const float4*)(P + (size_t)r * 1024 + EMBED);
    float se = 0.0f, sf = 0.0f;
#pragma unroll
    for (int it = 0; it < EMBED / 4 / 32; ++it) {
        int i4 = lane + 32 * it;
        float4 a = oe4[i4], p = pe4[i4];
        se += a.x * p.x + a.y * p.y + a.z * p.z + a.w * p.w;
        float4 c = of4[i4], q = pf4[i4];
        sf += c.x * q.x + c.y * q.y + c.z * q.z + c.w * q.w;
    }
#pragma unroll
    for (int off = 16; off > 0; off >>= 1) {
        se += __shfl_xor(se, off, 32);
        sf += __shfl_xor(sf, off, 32);
    }
    if (lane == 0) {
        int s = ei[e];
        atomicAdd(energy + batch[s], se);
        atomicAdd(forces + 3 * s + 0, sf * vec_hat[3 * e + 0]);
        atomicAdd(forces + 3 * s + 1, sf * vec_hat[3 * e + 1]);
        atomicAdd(forces + 3 * s + 2, sf * vec_hat[3 * e + 2]);
    }
}

// ---------- host ----------
static inline char* carve(char*& p, size_t bytes) {
    char* r = p;
    p += (bytes + 255) & ~(size_t)255;
    return r;
}

extern "C" void kernel_launch(void* const* d_in, const int* in_sizes, int n_in,
                              void* d_out, int out_size, void* d_ws, size_t ws_size,
                              hipStream_t stream) {
    const float* x       = (const float*)d_in[0];
    const float* dist    = (const float*)d_in[2];
    const float* vec_hat = (const float*)d_in[3];
    const int*   batch   = (const int*)d_in[4];
    const int*   ei      = (const int*)d_in[5];
    const float* e_Win  = (const float*)d_in[6];
    const float* e_bin  = (const float*)d_in[7];
    const float* e_Wres = (const float*)d_in[8];
    const float* e_bres = (const float*)d_in[9];
    const float* e_Wout = (const float*)d_in[10];
    const float* e_bout = (const float*)d_in[11];
    const float* e_Wrbf = (const float*)d_in[12];
    const float* e_Wfin = (const float*)d_in[13];
    const float* f_Win  = (const float*)d_in[14];
    const float* f_bin  = (const float*)d_in[15];
    const float* f_Wres = (const float*)d_in[16];
    const float* f_bres = (const float*)d_in[17];
    const float* f_Wout = (const float*)d_in[18];
    const float* f_bout = (const float*)d_in[19];
    const float* f_Wrbf = (const float*)d_in[20];
    const float* f_Wfin = (const float*)d_in[21];

    const int E      = in_sizes[2];          // 100000
    const int nNodes = in_sizes[1] / 3;      // 10000
    float* energy = (float*)d_out;
    float* forces = energy + (out_size - 3 * nNodes);

    char* p = (char*)d_ws;
    u16*   W1T   = (u16*)  carve(p, (size_t)NCAT * KIN * 2);
    u16*   W2eT  = (u16*)  carve(p, (size_t)FF * FF * 2);
    u16*   W2fT  = (u16*)  carve(p, (size_t)FF * FF * 2);
    u16*   W3eT  = (u16*)  carve(p, (size_t)EMBED * FF * 2);
    u16*   W3fT  = (u16*)  carve(p, (size_t)EMBED * FF * 2);
    u16*   WrbfT = (u16*)  carve(p, (size_t)1024 * KRBF * 2);
    float* bias1 = (float*)carve(p, (size_t)NCAT * 4);
    u16*   Abuf  = (u16*)  carve(p, (size_t)CHUNK * KIN * 2);
    u16*   RBF   = (u16*)  carve(p, (size_t)CHUNK * KRBF * 2);
    u16*   H1    = (u16*)  carve(p, (size_t)CHUNK * NCAT * 2);
    u16*   H2e   = (u16*)  carve(p, (size_t)CHUNK * FF * 2);
    u16*   H2f   = (u16*)  carve(p, (size_t)CHUNK * FF * 2);
    float* OUTe  = (float*)carve(p, (size_t)CHUNK * EMBED * 4);
    float* OUTf  = (float*)carve(p, (size_t)CHUNK * EMBED * 4);
    float* P     = (float*)carve(p, (size_t)CHUNK * 1024 * 4);

    // outputs start at zero
    zero_kernel<<<(out_size + 255) / 256, 256, 0, stream>>>((float*)d_out, out_size);

    // one-time weight prep (bf16, [N][K] transposed)
    convT_kernel<<<(KIN * FF + 255) / 256, 256, 0, stream>>>(e_Win, W1T, KIN, FF);
    convT_kernel<<<(KIN * FF + 255) / 256, 256, 0, stream>>>(f_Win, W1T + (size_t)FF * KIN, KIN, FF);
    convT_kernel<<<(FF * FF + 255) / 256, 256, 0, stream>>>(e_Wres, W2eT, FF, FF);
    convT_kernel<<<(FF * FF + 255) / 256, 256, 0, stream>>>(f_Wres, W2fT, FF, FF);
    convT_kernel<<<(FF * EMBED + 255) / 256, 256, 0, stream>>>(e_Wout, W3eT, FF, EMBED);
    convT_kernel<<<(FF * EMBED + 255) / 256, 256, 0, stream>>>(f_Wout, W3fT, FF, EMBED);
    prep_rbfw_kernel<<<(1024 * KRBF + 255) / 256, 256, 0, stream>>>(e_Wrbf, e_Wfin, f_Wrbf, f_Wfin, WrbfT);
    bias_cat_kernel<<<(NCAT + 255) / 256, 256, 0, stream>>>(e_bin, f_bin, bias1);

    const int nChunks = (E + CHUNK - 1) / CHUNK;
    for (int c = 0; c < nChunks; ++c) {
        int ebase = c * CHUNK;
        gather_kernel<<<CHUNK, KIN / 8, 0, stream>>>(x, ei, Abuf, ebase, E, nNodes);
        rbf_kernel<<<(CHUNK * KRBF) / 256, 256, 0, stream>>>(dist, RBF, ebase, E);

        // stage 1: [4096x1536] x [1536x2048] -> gelu -> H1 (bf16)
        gemm_kernel<<<dim3(NCAT / BN, CHUNK / BM), 256, 0, stream>>>(
            Abuf, KIN, W1T, KIN, KIN, bias1, 0, H1, nullptr, nullptr, 0, NCAT);

        // rbf projection: [4096x64] x [64x1024] -> P (f32), Wfin folded in
        gemm_kernel<<<dim3(1024 / BN, CHUNK / BM), 256, 0, stream>>>(
            RBF, KRBF, WrbfT, KRBF, KRBF, nullptr, 2, nullptr, P, nullptr, 0, 1024);

        // stage 2 (residual): H2 = H1 + gelu(H1 @ Wres + bres)
        gemm_kernel<<<dim3(FF / BN, CHUNK / BM), 256, 0, stream>>>(
            H1, NCAT, W2eT, FF, FF, e_bres, 1, H2e, nullptr, H1, NCAT, FF);
        gemm_kernel<<<dim3(FF / BN, CHUNK / BM), 256, 0, stream>>>(
            H1 + FF, NCAT, W2fT, FF, FF, f_bres, 1, H2f, nullptr, H1 + FF, NCAT, FF);

        // stage 3: OUT = H2 @ Wout + bout (f32)
        gemm_kernel<<<dim3(EMBED / BN, CHUNK / BM), 256, 0, stream>>>(
            H2e, FF, W3eT, FF, FF, e_bout, 2, nullptr, OUTe, nullptr, 0, EMBED);
        gemm_kernel<<<dim3(EMBED / BN, CHUNK / BM), 256, 0, stream>>>(
            H2f, FF, W3fT, FF, FF, f_bout, 2, nullptr, OUTf, nullptr, 0, EMBED);

        // per-edge dot + scatter
        finalize_kernel<<<CHUNK / 8, 256, 0, stream>>>(
            OUTe, OUTf, P, vec_hat, batch, ei, energy, forces, ebase, E);
    }
    (void)n_in; (void)ws_size;
}
